// NoTPAttention_20023137534549
// MI455X (gfx1250) — compile-verified, matrix operand order checked by reading, not run on hardware
//
#include <hip/hip_runtime.h>
#include <hip/hip_bf16.h>
#include <math.h>

#define HID   2048
#define NHEAD 16
#define HDIM  128
#define BATCH 2
#define SEQ   2048
#define MTOT  (BATCH*SEQ)      // 4096
#define NQKV  (3*HID)          // 6144

typedef __attribute__((ext_vector_type(16))) __bf16 v16bf;
typedef __attribute__((ext_vector_type(8)))  __bf16 v8bf;
typedef __attribute__((ext_vector_type(8)))  float  v8f;
typedef __attribute__((ext_vector_type(4)))  float  v4f;

static __device__ __forceinline__ v16bf cat8(v8bf lo, v8bf hi) {
    return __builtin_shufflevector(lo, hi, 0,1,2,3,4,5,6,7,8,9,10,11,12,13,14,15);
}

// Load one 16x32 bf16 WMMA fragment from a row-major matrix (ld elements/row).
// Lane sub (0..15) supplies row (row0+sub); half selects the documented K split:
//   half=0 -> K = {0..7, 16..23}, half=1 -> K = {8..15, 24..31}  (relative to k0)
static __device__ __forceinline__ v16bf load_frag(const __bf16* __restrict__ p,
                                                  int ld, int row0, int k0,
                                                  int sub, int half) {
    const __bf16* base = p + (size_t)(row0 + sub) * ld + k0 + half * 8;
    v8bf lo = *(const v8bf*)(base);
    v8bf hi = *(const v8bf*)(base + 16);
    return cat8(lo, hi);
}

__global__ void convert_f32_bf16(const float* __restrict__ in,
                                 __bf16* __restrict__ out, int n) {
    int i = blockIdx.x * blockDim.x + threadIdx.x;
    int stride = gridDim.x * blockDim.x;
    for (; i < n; i += stride) out[i] = (__bf16)in[i];
}

// Generic bf16 GEMM:  out[m][n] = sum_k A[m][k] * B[n][k]  (+ bias[n])
// Wave tile 64x64 = 4x4 WMMA tiles: 16 WMMAs per 8 fragments (0.5 KB operand
// traffic per WMMA) so the matrix pipes, not the cache path, are the limiter.
// Software-pipelined: fragments for k+32 load while WMMAs for k issue.
// Block = 256 threads = 8 waves (2x4); block tile 128x256.
// Epilogue: all Q/K/V routing decisions are WAVE-UNIFORM (the wave's 64-col
// span never crosses a 2048 Q/K/V boundary and each 16-col sub-tile stays
// inside one 128-col head), so selection is scalar-branch only.
// MODE 0: write fp32 to Cout (out-projection).
// MODE 1: QKV epilogue: scatter bf16 into Q[b,h,s,d], K[b,h,s,d], Vt[b,h,d,s].
template <int MODE>
__global__ __launch_bounds__(256)
void gemm_bf16(const __bf16* __restrict__ A, const __bf16* __restrict__ Bm,
               const float* __restrict__ bias, float* __restrict__ Cout,
               __bf16* __restrict__ Qb, __bf16* __restrict__ Kb,
               __bf16* __restrict__ Vt, int M, int N, int Kdim) {
    const int lane = threadIdx.x & 31;
    const int sub  = lane & 15;
    const int half = lane >> 4;
    const int wave = threadIdx.x >> 5;
    const int wm = wave & 1;            // 2 wave-rows
    const int wn = wave >> 1;           // 4 wave-cols
    const int m0 = blockIdx.x * 128 + wm * 64;
    const int n0 = blockIdx.y * 256 + wn * 64;

    v8f c[4][4] = {};

    v16bf a[4], bfr[4];
#pragma unroll
    for (int i = 0; i < 4; i++) {
        a[i]   = load_frag(A,  Kdim, m0 + 16 * i, 0, sub, half);
        bfr[i] = load_frag(Bm, Kdim, n0 + 16 * i, 0, sub, half);
    }

    for (int k0 = 0; k0 < Kdim - 32; k0 += 32) {
        const int kn = k0 + 32;
        // next chunk's loads issue first and overlap this chunk's 16 WMMAs
        v16bf na[4], nb[4];
#pragma unroll
        for (int i = 0; i < 4; i++) {
            na[i] = load_frag(A,  Kdim, m0 + 16 * i, kn, sub, half);
            nb[i] = load_frag(Bm, Kdim, n0 + 16 * i, kn, sub, half);
        }
        __builtin_prefetch(A  + (size_t)(m0 + sub) * Kdim + kn + 32, 0, 3);
        __builtin_prefetch(Bm + (size_t)(n0 + sub) * Kdim + kn + 32, 0, 3);
#pragma unroll
        for (int i = 0; i < 4; i++)
#pragma unroll
            for (int j = 0; j < 4; j++)
                c[i][j] = __builtin_amdgcn_wmma_f32_16x16x32_bf16(
                    false, a[i], false, bfr[j], (short)0, c[i][j], false, false);
#pragma unroll
        for (int i = 0; i < 4; i++) { a[i] = na[i]; bfr[i] = nb[i]; }
    }
#pragma unroll
    for (int i = 0; i < 4; i++)
#pragma unroll
        for (int j = 0; j < 4; j++)
            c[i][j] = __builtin_amdgcn_wmma_f32_16x16x32_bf16(
                false, a[i], false, bfr[j], (short)0, c[i][j], false, false);

    // ---- epilogue ----
    // C/D layout: VGPR j holds row (j + 8*half), column = sub.
    const int bidx  = m0 >> 11;              // uniform: batch index
    const int sbase = m0 & (SEQ - 1);        // uniform: seq base of this tile

#pragma unroll
    for (int tj = 0; tj < 4; tj++) {
        const int colbase = n0 + tj * 16;    // uniform
        const float bcol  = bias[colbase + sub];
        if (MODE == 0) {
#pragma unroll
            for (int ti = 0; ti < 4; ti++)
#pragma unroll
                for (int j = 0; j < 8; j++) {
                    const int row = m0 + ti * 16 + j + 8 * half;
                    Cout[(size_t)row * N + colbase + sub] = c[ti][tj][j] + bcol;
                }
        } else {
            const int t     = colbase >> 11;             // uniform: 0=Q 1=K 2=V
            const int wbase = colbase & (HID - 1);       // uniform
            const int h     = wbase >> 7;                // uniform head
            const int d     = (wbase & (HDIM - 1)) + sub; // per-lane d
            const size_t bh = (size_t)(bidx * NHEAD + h);
            if (t < 2) {   // Q or K: [b,h,s,d], d contiguous across lanes
                __bf16* dst = (t == 0 ? Qb : Kb) + bh * SEQ * HDIM + d;
#pragma unroll
                for (int ti = 0; ti < 4; ti++)
#pragma unroll
                    for (int j = 0; j < 8; j++) {
                        const int sidx = sbase + ti * 16 + j + 8 * half;
                        dst[(size_t)sidx * HDIM] = (__bf16)(c[ti][tj][j] + bcol);
                    }
            } else {       // V transposed: [b,h,d,s]
                __bf16* dst = Vt + (bh * HDIM + d) * SEQ + sbase;
#pragma unroll
                for (int ti = 0; ti < 4; ti++)
#pragma unroll
                    for (int j = 0; j < 8; j++)
                        dst[ti * 16 + j + 8 * half] = (__bf16)(c[ti][tj][j] + bcol);
            }
        }
    }
}

// Flash attention: one wave per 16-query tile; 4 waves / block.
// Scores: 8 batched K-frag loads then 8 WMMAs. Softmax done in TRANSPOSED
// layout: raw scores go through LDS once; each lane then owns one row
// (16 of its 32 columns; partner lane ^16 owns the rest), so max/sum need a
// single __shfl_xor(.,16) each and m/l are scalar per lane (no reg arrays).
// The exp'd row IS the A-fragment for P*V (lane=row, ISA half-split K order).
__global__ __launch_bounds__(128)
void attn_kernel(const __bf16* __restrict__ Qb, const __bf16* __restrict__ Kb,
                 const __bf16* __restrict__ Vt, __bf16* __restrict__ attn) {
    __shared__ float psh[4][16 * 32];     // per-wave 16x32 f32 score tile

    const int lane = threadIdx.x & 31;
    const int sub  = lane & 15;
    const int half = lane >> 4;
    const int wave = threadIdx.x >> 5;
    const int h = blockIdx.y, b = blockIdx.z;
    const int s0 = blockIdx.x * 64 + wave * 16;
    const size_t bh = (size_t)(b * NHEAD + h);
    const __bf16* Qp = Qb + bh * SEQ * HDIM;
    const __bf16* Kp = Kb + bh * SEQ * HDIM;
    const __bf16* Vp = Vt + bh * HDIM * SEQ;

    v16bf q[4];
#pragma unroll
    for (int cc = 0; cc < 4; cc++) q[cc] = load_frag(Qp, HDIM, s0, cc * 32, sub, half);

    v8f   o[8] = {};
    float m_i = -INFINITY, l_i = 0.f;
    // scores kept in log2 domain: Cs = (1/sqrt(128)) * log2(e)
    const float Cs = 0.08838834764831845f * 1.4426950408889634f;

    for (int k0 = 0; k0 < SEQ; k0 += 32) {
        // ---- scores: batch all 8 fragment loads, then 8 WMMAs ----
        v16bf kb[8];
#pragma unroll
        for (int cc = 0; cc < 4; cc++) {
            kb[2*cc]   = load_frag(Kp, HDIM, k0,      cc * 32, sub, half);
            kb[2*cc+1] = load_frag(Kp, HDIM, k0 + 16, cc * 32, sub, half);
        }
        __builtin_prefetch(Kp + (size_t)(k0 + 32 + sub) * HDIM, 0, 3);
        v8f sc0 = {}, sc1 = {};
#pragma unroll
        for (int cc = 0; cc < 4; cc++) {
            sc0 = __builtin_amdgcn_wmma_f32_16x16x32_bf16(false, q[cc], false, kb[2*cc],   (short)0, sc0, false, false);
            sc1 = __builtin_amdgcn_wmma_f32_16x16x32_bf16(false, q[cc], false, kb[2*cc+1], (short)0, sc1, false, false);
        }

        // ---- C-layout -> row-major via LDS (single pass) ----
        float* lp = psh[wave];
#pragma unroll
        for (int j = 0; j < 8; j++) {
            lp[(j + 8 * half) * 32 + sub]      = sc0[j];
            lp[(j + 8 * half) * 32 + 16 + sub] = sc1[j];
        }
        asm volatile("" ::: "memory");     // same-wave DS ops are in-order
        const float* rp = lp + sub * 32 + half * 8;   // this lane: row=sub, its 16 cols
        v4f x0 = *(const v4f*)(rp);
        v4f x1 = *(const v4f*)(rp + 4);
        v4f y0 = *(const v4f*)(rp + 16);
        v4f y1 = *(const v4f*)(rp + 20);
        asm volatile("" ::: "memory");

        float xs[16];
#pragma unroll
        for (int i = 0; i < 4; i++) {
            xs[i]      = x0[i] * Cs;
            xs[4 + i]  = x1[i] * Cs;
            xs[8 + i]  = y0[i] * Cs;
            xs[12 + i] = y1[i] * Cs;
        }
        // ---- per-lane online softmax (row = sub) ----
        float mx = xs[0];
#pragma unroll
        for (int i = 1; i < 16; i++) mx = fmaxf(mx, xs[i]);
        mx = fmaxf(mx, __shfl_xor(mx, 16));          // merge partner half-row
        const float mnew = fmaxf(m_i, mx);
        const float alpha = exp2f(m_i - mnew);
        float sum = 0.f;
        v16bf pf;
#pragma unroll
        for (int i = 0; i < 16; i++) {
            const float p = exp2f(xs[i] - mnew);
            sum += p;
            pf[i] = (__bf16)p;                        // exactly the P A-fragment
        }
        sum += __shfl_xor(sum, 16);
        l_i = l_i * alpha + sum;
        m_i = mnew;

        // ---- broadcast alpha back to C layout and rescale accumulators ----
        float aj[8];
#pragma unroll
        for (int j = 0; j < 8; j++) aj[j] = __shfl(alpha, 8 * half + j, 16);
#pragma unroll
        for (int n = 0; n < 8; n++)
#pragma unroll
            for (int j = 0; j < 8; j++) o[n][j] *= aj[j];

        // ---- P*V: batch all 8 V-frag loads, then 8 WMMAs ----
        v16bf vf[8];
#pragma unroll
        for (int n = 0; n < 8; n++) vf[n] = load_frag(Vp, SEQ, n * 16, k0, sub, half);
        __builtin_prefetch(Vp + (size_t)sub * SEQ + k0 + 32, 0, 3);
#pragma unroll
        for (int n = 0; n < 8; n++)
            o[n] = __builtin_amdgcn_wmma_f32_16x16x32_bf16(false, pf, false, vf[n], (short)0, o[n], false, false);
    }

    // ---- epilogue: broadcast l, normalize, store ----
    float inv_l[8];
#pragma unroll
    for (int j = 0; j < 8; j++) inv_l[j] = 1.0f / __shfl(l_i, 8 * half + j, 16);
#pragma unroll
    for (int n = 0; n < 8; n++) {
#pragma unroll
        for (int j = 0; j < 8; j++) {
            const int row = b * SEQ + s0 + j + 8 * half;
            const int col = h * HDIM + n * 16 + sub;
            attn[(size_t)row * HID + col] = (__bf16)(o[n][j] * inv_l[j]);
        }
    }
}

extern "C" void kernel_launch(void* const* d_in, const int* in_sizes, int n_in,
                              void* d_out, int out_size, void* d_ws, size_t ws_size,
                              hipStream_t stream) {
    (void)in_sizes; (void)n_in; (void)out_size; (void)ws_size;
    const float* x     = (const float*)d_in[0];
    const float* w_qkv = (const float*)d_in[1];
    const float* b_qkv = (const float*)d_in[2];
    const float* w_out = (const float*)d_in[3];
    const float* b_out = (const float*)d_in[4];
    float* out = (float*)d_out;

    char* ws = (char*)d_ws;
    size_t off = 0;
    auto alloc = [&](size_t bytes) -> void* {
        void* p = ws + off;
        off += (bytes + 255) & ~(size_t)255;
        return p;
    };
    __bf16* xb    = (__bf16*)alloc((size_t)MTOT * HID * 2);
    __bf16* wqkvb = (__bf16*)alloc((size_t)NQKV * HID * 2);
    __bf16* wob   = (__bf16*)alloc((size_t)HID * HID * 2);
    __bf16* Qb    = (__bf16*)alloc((size_t)BATCH * NHEAD * SEQ * HDIM * 2);
    __bf16* Kb    = (__bf16*)alloc((size_t)BATCH * NHEAD * SEQ * HDIM * 2);
    __bf16* Vt    = (__bf16*)alloc((size_t)BATCH * NHEAD * SEQ * HDIM * 2);
    __bf16* attn  = (__bf16*)alloc((size_t)MTOT * HID * 2);

    convert_f32_bf16<<<2048, 256, 0, stream>>>(x,     xb,    MTOT * HID);
    convert_f32_bf16<<<2048, 256, 0, stream>>>(w_qkv, wqkvb, NQKV * HID);
    convert_f32_bf16<<<2048, 256, 0, stream>>>(w_out, wob,   HID * HID);

    dim3 g1(MTOT / 128, NQKV / 256);
    gemm_bf16<1><<<g1, 256, 0, stream>>>(xb, wqkvb, b_qkv, nullptr,
                                         Qb, Kb, Vt, MTOT, NQKV, HID);

    dim3 g2(SEQ / 64, NHEAD, BATCH);
    attn_kernel<<<g2, 128, 0, stream>>>(Qb, Kb, Vt, attn);

    dim3 g3(MTOT / 128, HID / 256);
    gemm_bf16<0><<<g3, 256, 0, stream>>>(attn, wob, b_out, out,
                                         nullptr, nullptr, nullptr, MTOT, HID, HID);
}
